// TopKSAE_29008209117485
// MI455X (gfx1250) — matrix operand order repaired and checked by reading, not compiled, not run on hardware
//
#include <hip/hip_runtime.h>
#include <hip/hip_bf16.h>

// ---------------------------------------------------------------------------
// TopK SAE forward for MI455X (gfx1250, wave32, WMMA, async-to-LDS staging).
//   z      = (x - b_pre) @ W_enc^T + b_enc          [8192, 24576]  (bf16 WMMA, f32 accum)
//   topk   = exact top-64 per row (radix select)     -> z_sparse (in place)
//   recon  = z_sparse @ W_dec^T + b_dec              (sparse: 64 gathers/row)
// Outputs concatenated: recon [8192*768] then z_sparse [8192*24576].
// ---------------------------------------------------------------------------

#define D_MODEL 768
#define D_SAE   24576
#define TOPK    64
#define BATCH   8192

// LDS tile geometry: block tile 128(M) x 128(N), k-step 32, padded row stride
// of 40 bf16 (80 B -> 20-bank step: conflict-free ds_load_b128 across 16 rows,
// rows stay 16B-aligned).
#define BK       32
#define LDS_STRD 40

typedef __attribute__((ext_vector_type(16))) __bf16 v16bf;
typedef __attribute__((ext_vector_type(8)))  __bf16 v8bf;
typedef __attribute__((ext_vector_type(8)))  float  v8f;

// ------------------------------ f32 -> bf16 conversion ---------------------

__global__ __launch_bounds__(256) void sae_cvt_x(const float* __restrict__ x,
                                                 const float* __restrict__ b_pre,
                                                 __bf16* __restrict__ out) {
    int e = blockIdx.x * 256 + threadIdx.x;
    if (e < BATCH * D_MODEL) {
        float v = x[e] - b_pre[e % D_MODEL];
        out[e] = (__bf16)v;
    }
}

__global__ __launch_bounds__(256) void sae_cvt_w(const float* __restrict__ w,
                                                 __bf16* __restrict__ out) {
    int e = blockIdx.x * 256 + threadIdx.x;
    if (e < D_SAE * D_MODEL) {
        out[e] = (__bf16)w[e];
    }
}

// ------------------------------ async copy helpers -------------------------
// GLOBAL_LOAD_ASYNC_TO_LDS_B128: memory -> LDS, no VGPR staging, ASYNCcnt.

__device__ inline void async_copy16(unsigned lds_addr, unsigned long long gaddr) {
    asm volatile("global_load_async_to_lds_b128 %0, %1, off"
                 :: "v"(lds_addr), "v"(gaddr) : "memory");
}

__device__ inline void wait_async0() {
    asm volatile("s_wait_asynccnt 0x0" ::: "memory");
}

__device__ inline unsigned lds_off(const void* p) {
    // __shared__ pointers cast to generic carry the LDS byte offset in the
    // low 32 bits (aperture in the high bits) -- truncation yields LDS addr.
    return (unsigned)(unsigned long long)(uintptr_t)p;
}

// ------------------------------ fragment loads -----------------------------
// A frag (16-bit A 16x32, ISA 7.12.2): lane m = lane&15, K chunks
//   {kb..kb+7} and {16+kb..16+kb+7}, kb = (lane>>4)*8  -> two 16B loads 16B apart*2.
// B frag (16-bit B 32x16): lane n = lane&15, K chunks as for A (loaded via fragA at kb)
//   -> two 16B loads 32B apart, same k order as A.

__device__ inline v16bf fragA(const __bf16* p) {   // p at (row, kb)
    v8bf lo = *(const v8bf*)(p);
    v8bf hi = *(const v8bf*)(p + 16);
    return __builtin_shufflevector(lo, hi, 0,1,2,3,4,5,6,7,8,9,10,11,12,13,14,15);
}

__device__ inline v16bf fragB(const __bf16* p) {   // p at (row, kc)
    v8bf lo = *(const v8bf*)(p);
    v8bf hi = *(const v8bf*)(p + 8);
    return __builtin_shufflevector(lo, hi, 0,1,2,3,4,5,6,7,8,9,10,11,12,13,14,15);
}

// ------------------------------ encode GEMM (WMMA) -------------------------
// A: [BATCH, 768] bf16 row-major (x - b_pre)
// B: [D_SAE, 768] bf16 row-major (W_enc rows; acts as B^T)
// Z: [BATCH, D_SAE] f32 dense, written into z_sparse output region.
// Block: 256 threads (8 wave32), tile 128x128; wave tile 32(M)x64(N) = 8 WMMA.

__global__ __launch_bounds__(256) void sae_encode_wmma(const __bf16* __restrict__ A,
                                                       const __bf16* __restrict__ B,
                                                       const float* __restrict__ b_enc,
                                                       float* __restrict__ Z) {
    __shared__ __attribute__((aligned(16))) __bf16 AsLds[2][128 * LDS_STRD];
    __shared__ __attribute__((aligned(16))) __bf16 BsLds[2][128 * LDS_STRD];

    const int tid  = threadIdx.x;
    const int lane = tid & 31;
    const int wave = tid >> 5;          // 0..7
    const int wm   = wave & 3;          // 4 waves along M (32 rows each)
    const int wn   = wave >> 2;         // 2 waves along N (64 cols each)

    const int m_block = blockIdx.y * 128;
    const int n_block = blockIdx.x * 128;

    // copy assignment: 2 threads per row, 32B (2 x b128) each, per matrix
    const int tr = tid >> 1;            // 0..127
    const int tc = (tid & 1) * 16;      // element offset 0 or 16

    const unsigned a_lds[2] = { lds_off(&AsLds[0][tr * LDS_STRD + tc]),
                                lds_off(&AsLds[1][tr * LDS_STRD + tc]) };
    const unsigned b_lds[2] = { lds_off(&BsLds[0][tr * LDS_STRD + tc]),
                                lds_off(&BsLds[1][tr * LDS_STRD + tc]) };
    const __bf16* a_g = A + (size_t)(m_block + tr) * D_MODEL + tc;
    const __bf16* b_g = B + (size_t)(n_block + tr) * D_MODEL + tc;

    auto issue_copy = [&](int buf, int kk) {
        unsigned long long ga = (unsigned long long)(uintptr_t)(a_g + kk);
        unsigned long long gb = (unsigned long long)(uintptr_t)(b_g + kk);
        async_copy16(a_lds[buf],      ga);
        async_copy16(a_lds[buf] + 16, ga + 16);
        async_copy16(b_lds[buf],      gb);
        async_copy16(b_lds[buf] + 16, gb + 16);
    };

    const int kb = (lane >> 4) << 3;    // A-frag K sub-offset
    const int kc = (lane >> 4) << 4;    // unused; B frag loads at kb
    const int rl = lane & 15;

    v8f c[2][4] = {};

    auto compute = [&](int buf) {
        v16bf a[2], b[4];
        #pragma unroll
        for (int tm = 0; tm < 2; tm++)
            a[tm] = fragA(&AsLds[buf][(wm * 32 + tm * 16 + rl) * LDS_STRD + kb]);
        #pragma unroll
        for (int tn = 0; tn < 4; tn++)
            b[tn] = fragA(&BsLds[buf][(wn * 64 + tn * 16 + rl) * LDS_STRD + kb]);
        #pragma unroll
        for (int tm = 0; tm < 2; tm++)
            #pragma unroll
            for (int tn = 0; tn < 4; tn++)
                c[tm][tn] = __builtin_amdgcn_wmma_f32_16x16x32_bf16(
                    false, a[tm], false, b[tn], (short)0, c[tm][tn], false, false);
    };

    // ---- software pipeline: double-buffered async fills ----
    issue_copy(0, 0);
    wait_async0();
    __syncthreads();

    for (int kk = 0; kk < D_MODEL; kk += 2 * BK) {
        if (kk + BK < D_MODEL) issue_copy(1, kk + BK);
        compute(0);
        wait_async0();
        __syncthreads();

        if (kk + 2 * BK < D_MODEL) issue_copy(0, kk + 2 * BK);
        compute(1);
        wait_async0();
        __syncthreads();
    }

    // ---- epilogue: C/D layout VGPR r -> m = r + ((lane>=16)?8:0), n = lane&15
    const int mofs = (lane >> 4) << 3;
    #pragma unroll
    for (int tm = 0; tm < 2; tm++) {
        #pragma unroll
        for (int tn = 0; tn < 4; tn++) {
            int nbase = n_block + wn * 64 + tn * 16 + rl;
            float bias = b_enc[nbase];
            #pragma unroll
            for (int r = 0; r < 8; r++) {
                int m = m_block + wm * 32 + tm * 16 + r + mofs;
                Z[(size_t)m * D_SAE + nbase] = c[tm][tn][r] + bias;
            }
        }
    }
}

// ------------------------------ top-k radix select -------------------------
// One 256-thread block per row. Exact, deterministic. Works in place on the
// dense z stored in the z_sparse output region; emits exactly 64 (val, idx)
// pairs per row into workspace (deterministic order).

__device__ inline unsigned f2key(float f) {
    unsigned b = __float_as_uint(f);
    return (b & 0x80000000u) ? ~b : (b | 0x80000000u);  // monotonic
}

__global__ __launch_bounds__(256) void sae_topk(float* __restrict__ ZS,
                                                float* __restrict__ vlist,
                                                int*   __restrict__ ilist) {
    const int row = blockIdx.x;
    const int tid = threadIdx.x;
    float* __restrict__ zr = ZS + (size_t)row * D_SAE;

    __shared__ int hist[256];
    __shared__ int cntG_s[256], cntE_s[256];
    __shared__ int sh_sel, sh_above, sh_totalG;

    // ---- 4-level byte radix select: find exact 64th-largest key ----
    unsigned prefix = 0, prefmask = 0;
    int remaining = TOPK;
    for (int shift = 24; shift >= 0; shift -= 8) {
        hist[tid] = 0;
        __syncthreads();
        for (int e = tid; e < D_SAE; e += 256) {
            unsigned key = f2key(zr[e]);
            if ((key & prefmask) == prefix)
                atomicAdd(&hist[(key >> shift) & 255], 1);
        }
        __syncthreads();
        if (tid == 0) {
            int acc = 0, sel = 0;
            for (int bi = 255; bi >= 0; bi--) {
                int c = hist[bi];
                if (acc + c >= remaining) { sel = bi; break; }
                acc += c;
            }
            sh_sel = sel; sh_above = acc;
        }
        __syncthreads();
        prefix   |= ((unsigned)sh_sel) << shift;
        prefmask |= 0xFFu << shift;
        remaining -= sh_above;
        __syncthreads();
    }
    const unsigned T = prefix;          // exact key of K-th largest

    // ---- deterministic compaction: count, scan, write ----
    int cg = 0, ce = 0;
    for (int e = tid; e < D_SAE; e += 256) {
        unsigned key = f2key(zr[e]);
        if (key > T) cg++;
        else if (key == T) ce++;
    }
    cntG_s[tid] = cg; cntE_s[tid] = ce;
    __syncthreads();
    if (tid == 0) {
        int acc = 0;
        for (int i = 0; i < 256; i++) { int t = cntG_s[i]; cntG_s[i] = acc; acc += t; }
        sh_totalG = acc;
        acc = 0;
        for (int i = 0; i < 256; i++) { int t = cntE_s[i]; cntE_s[i] = acc; acc += t; }
    }
    __syncthreads();

    int baseG = cntG_s[tid];
    int baseE = sh_totalG + cntE_s[tid];
    float* vrow = vlist + (size_t)row * TOPK;
    int*   irow = ilist + (size_t)row * TOPK;

    for (int e = tid; e < D_SAE; e += 256) {
        float val = zr[e];
        unsigned key = f2key(val);
        if (key > T) {
            vrow[baseG] = val; irow[baseG] = e; baseG++;   // value stays in place
        } else if (key == T) {
            if (baseE < TOPK) { vrow[baseE] = val; irow[baseE] = e; }
            else              { zr[e] = 0.0f; }
            baseE++;
        } else {
            zr[e] = 0.0f;
        }
    }
}

// ------------------------------ sparse decode ------------------------------
// recon[b, d] = b_dec[d] + sum_k v[b,k] * W_dec[d, idx[b,k]]
// W_dec (75.5 MB) is L2-resident; fixed summation order => deterministic.

__global__ __launch_bounds__(256) void sae_decode(const float* __restrict__ W_dec,
                                                  const float* __restrict__ b_dec,
                                                  const float* __restrict__ vlist,
                                                  const int*   __restrict__ ilist,
                                                  float* __restrict__ recon) {
    const int row = blockIdx.x;
    const int tid = threadIdx.x;
    __shared__ float sv[TOPK];
    __shared__ int   si[TOPK];
    if (tid < TOPK) {
        sv[tid] = vlist[(size_t)row * TOPK + tid];
        si[tid] = ilist[(size_t)row * TOPK + tid];
    }
    __syncthreads();

    for (int d = tid; d < D_MODEL; d += 256) {
        const float* wrow = W_dec + (size_t)d * D_SAE;
        float acc = b_dec[d];
        #pragma unroll 8
        for (int k = 0; k < TOPK; k++)
            acc += sv[k] * wrow[si[k]];
        recon[(size_t)row * D_MODEL + d] = acc;
    }
}

// ------------------------------ launcher -----------------------------------

extern "C" void kernel_launch(void* const* d_in, const int* in_sizes, int n_in,
                              void* d_out, int out_size, void* d_ws, size_t ws_size,
                              hipStream_t stream) {
    const float* x     = (const float*)d_in[0];
    const float* W_enc = (const float*)d_in[1];
    const float* b_enc = (const float*)d_in[2];
    const float* W_dec = (const float*)d_in[3];
    const float* b_dec = (const float*)d_in[4];
    const float* b_pre = (const float*)d_in[5];

    float* out   = (float*)d_out;
    float* recon = out;                                   // [8192, 768]
    float* zs    = out + (size_t)BATCH * D_MODEL;         // [8192, 24576]

    // workspace layout
    char* ws = (char*)d_ws;
    __bf16* Abf  = (__bf16*)(ws);                                              // 12.6 MB
    __bf16* Bbf  = (__bf16*)(ws + (size_t)BATCH * D_MODEL * 2);                // 37.7 MB
    float*  vlst = (float*) (ws + (size_t)(BATCH + D_SAE) * D_MODEL * 2);      // 2 MB
    int*    ilst = (int*)   (ws + (size_t)(BATCH + D_SAE) * D_MODEL * 2
                                + (size_t)BATCH * TOPK * 4);                   // 2 MB

    // 1) convert operands to bf16 (x - b_pre folded in)
    sae_cvt_x<<<(BATCH * D_MODEL + 255) / 256, 256, 0, stream>>>(x, b_pre, Abf);
    sae_cvt_w<<<(D_SAE * D_MODEL + 255) / 256, 256, 0, stream>>>(W_enc, Bbf);

    // 2) encode GEMM: dense z into z_sparse region
    dim3 ggrid(D_SAE / 128, BATCH / 128);
    sae_encode_wmma<<<ggrid, 256, 0, stream>>>(Abf, Bbf, b_enc, zs);

    // 3) exact top-64 per row, in place + compact lists
    sae_topk<<<BATCH, 256, 0, stream>>>(zs, vlst, ilst);

    // 4) sparse decode
    sae_decode<<<BATCH, 256, 0, stream>>>(W_dec, b_dec, vlst, ilst, recon);
}
